// GraphTransformerLayer_21569325760860
// MI455X (gfx1250) — matrix operand order repaired and checked by reading, not compiled, not run on hardware
//
#include <hip/hip_runtime.h>
#include <hip/hip_bf16.h>
#include <math.h>

// ---------------------------------------------------------------------------
// Types / WMMA helpers (CDNA5 / gfx1250, wave32)
// ---------------------------------------------------------------------------
typedef __attribute__((ext_vector_type(16))) _Float16 v16h;
typedef __attribute__((ext_vector_type(8)))  float    v8f;
typedef __attribute__((ext_vector_type(4)))  float    f4;
typedef __attribute__((ext_vector_type(4)))  _Float16 h4;

#define FLAG_BIAS 1
#define FLAG_RES  2
#define FLAG_LN   8

__device__ __forceinline__ v8f wmma16(v16h a, v16h b, v8f c) {
  return __builtin_amdgcn_wmma_f32_16x16x32_f16(
      false, a, false, b, (short)0, c, false, false);
}

// A fragment (16x32 f16) per ISA 7.12.2 wave32 layout.
__device__ __forceinline__ v16h load_a_frag(const _Float16* p, int lda, int lane) {
  const _Float16* r = p + (lane & 15) * lda;
  const int half8 = (lane >> 4) << 3;
  v16h f;
#pragma unroll
  for (int g = 0; g < 8; ++g) {
    int k = ((g >> 2) << 4) + half8 + ((g & 3) << 1);
    f[2 * g]     = r[k];
    f[2 * g + 1] = r[k + 1];
  }
  return f;
}

// B fragment (32x16 f16): LDS layout col-major Bs[col][k] -> same k order as A fragment.
__device__ __forceinline__ v16h load_b_frag(const _Float16* p, int ldb, int lane) {
  const _Float16* c = p + (lane & 15) * ldb + ((lane >> 4) << 3);
  v16h f;
#pragma unroll
  for (int g = 0; g < 8; ++g) {
    int k = ((g >> 2) << 4) + ((g & 3) << 1);
    f[2 * g]     = c[k];
    f[2 * g + 1] = c[k + 1];
  }
  return f;
}

// ---------------------------------------------------------------------------
__global__ void zero_f32_kernel(float* __restrict__ p, long long n) {
  long long i = (long long)blockIdx.x * blockDim.x + threadIdx.x;
  long long stride = (long long)gridDim.x * blockDim.x;
  for (; i < n; i += stride) p[i] = 0.0f;
}

// ---------------------------------------------------------------------------
// Tiled WMMA GEMM, 128x128 tile, BK=32, 8 waves (4x2), 32x64 per wave.
// ---------------------------------------------------------------------------
#define LDA_S 40
#define LDB_S 40
#define LDC_S 132
#define LDT_S 264
#define GEMM_SMEM_BYTES ((128 * LDA_S + 128 * LDB_S) * 2 + 128 * LDC_S * 4)
#define FFN_SMEM_BYTES  ((128 * LDA_S + 128 * LDB_S) * 2 + 128 * LDC_S * 4)

// stage a 128x32 f32 tile (rows clamped, OOB zeroed) into f16 LDS [row][k]
__device__ __forceinline__ void stage_a_tile(const float* __restrict__ A,
                                             _Float16* As, int row0, int k0,
                                             int M, int Kdim, int tid) {
  const f4 zf = {0.0f, 0.0f, 0.0f, 0.0f};
#pragma unroll
  for (int i = 0; i < 4; ++i) {
    int e  = i * 256 + tid;        // 1024 float4 = 128 rows x 8
    int r  = e >> 3;
    int c4 = (e & 7) << 2;
    int gr = row0 + r;
    int cr = gr < M ? gr : M - 1;  // clamp: unconditional coalesced load
    f4 v = *(const f4*)(A + (size_t)cr * Kdim + k0 + c4);
    if (gr >= M) v = zf;           // lowers to cndmask, no branch
    *(h4*)(&As[r * LDA_S + c4]) = __builtin_convertvector(v, h4);
  }
}

// stage a 32-row f32 B slice transposed into f16 LDS [col][k]
__device__ __forceinline__ void stage_b_tile(const float* __restrict__ Brow0,
                                             _Float16* Bs, int rowStride,
                                             int tid) {
#pragma unroll
  for (int i = 0; i < 4; ++i) {
    int e  = i * 256 + tid;        // 1024 float4 = 32 k x 32
    int kk = e >> 5;
    int c4 = (e & 31) << 2;
    f4 v = *(const f4*)(Brow0 + (size_t)kk * rowStride + c4);
    h4 hv = __builtin_convertvector(v, h4);
    Bs[(c4 + 0) * LDB_S + kk] = hv.x;
    Bs[(c4 + 1) * LDB_S + kk] = hv.y;
    Bs[(c4 + 2) * LDB_S + kk] = hv.z;
    Bs[(c4 + 3) * LDB_S + kk] = hv.w;
  }
}

template <int FLAGS>
__global__ __launch_bounds__(256, 1)
void gemm_wmma_kernel(const float* __restrict__ A,
                      const float* __restrict__ B,
                      const float* __restrict__ bias,
                      const float* __restrict__ residual,
                      const float* __restrict__ gamma,
                      const float* __restrict__ beta,
                      float* __restrict__ out,
                      int M, int Kdim, int Ncols) {
  extern __shared__ char smem[];
  _Float16* As = (_Float16*)smem;                 // [128][LDA_S]
  _Float16* Bs = As + 128 * LDA_S;                // [128][LDB_S] col-major
  float*    Cs = (float*)(Bs + 128 * LDB_S);      // [128][LDC_S]

  const int tid  = threadIdx.x;
  const int lane = tid & 31;
  const int wave = tid >> 5;
  const int wm   = wave & 3;
  const int wn   = wave >> 2;
  const int row0 = blockIdx.x * 128;
  const int col0 = blockIdx.y * 128;

  v8f acc[2][4];
#pragma unroll
  for (int mt = 0; mt < 2; ++mt)
#pragma unroll
    for (int nt = 0; nt < 4; ++nt)
#pragma unroll
      for (int r = 0; r < 8; ++r) acc[mt][nt][r] = 0.0f;

  for (int k0 = 0; k0 < Kdim; k0 += 32) {
    stage_a_tile(A, As, row0, k0, M, Kdim, tid);
    stage_b_tile(B + (size_t)k0 * Ncols + col0, Bs, Ncols, tid);
    __syncthreads();

    v16h af[2], bf[4];
#pragma unroll
    for (int mt = 0; mt < 2; ++mt)
      af[mt] = load_a_frag(As + (wm * 32 + mt * 16) * LDA_S, LDA_S, lane);
#pragma unroll
    for (int nt = 0; nt < 4; ++nt)
      bf[nt] = load_b_frag(Bs + (wn * 64 + nt * 16) * LDB_S, LDB_S, lane);
#pragma unroll
    for (int mt = 0; mt < 2; ++mt)
#pragma unroll
      for (int nt = 0; nt < 4; ++nt)
        acc[mt][nt] = wmma16(af[mt], bf[nt], acc[mt][nt]);
    __syncthreads();
  }

  // dump accumulators (C/D layout: VGPR r -> row r + 8*(lane>>4), col lane&15)
  {
    const int halfsel = lane >> 4;
    const int cl = lane & 15;
#pragma unroll
    for (int mt = 0; mt < 2; ++mt)
#pragma unroll
      for (int nt = 0; nt < 4; ++nt) {
        int rb = wm * 32 + mt * 16 + halfsel * 8;
        int cb = wn * 64 + nt * 16 + cl;
#pragma unroll
        for (int r = 0; r < 8; ++r) Cs[(rb + r) * LDC_S + cb] = acc[mt][nt][r];
      }
  }
  __syncthreads();

  // epilogue: one thread per row, vectorized float4
  if (tid < 128) {
    int gr = row0 + tid;
    if (gr < M) {
      const f4* crow = (const f4*)(Cs + tid * LDC_S);
      const f4* brow = (const f4*)(bias + col0);
      const f4* rrow = (const f4*)(residual + (size_t)gr * Ncols + col0);
      f4*       orow = (f4*)(out + (size_t)gr * Ncols + col0);

      float mean = 0.0f, rs = 0.0f;
      if (FLAGS & FLAG_LN) {
        float s = 0.0f, s2 = 0.0f;
#pragma unroll 4
        for (int jv = 0; jv < 32; ++jv) {
          f4 v = crow[jv];
          if (FLAGS & FLAG_BIAS) v += brow[jv];
          if (FLAGS & FLAG_RES)  v += rrow[jv];
          s  += v.x + v.y + v.z + v.w;
          s2 += v.x * v.x + v.y * v.y + v.z * v.z + v.w * v.w;
        }
        mean = s * (1.0f / 128.0f);
        float var = s2 * (1.0f / 128.0f) - mean * mean;
        rs = rsqrtf(var + 1e-5f);
      }
#pragma unroll 4
      for (int jv = 0; jv < 32; ++jv) {
        f4 v = crow[jv];
        if (FLAGS & FLAG_BIAS) v += brow[jv];
        if (FLAGS & FLAG_RES)  v += rrow[jv];
        if (FLAGS & FLAG_LN) {
          f4 g = ((const f4*)(gamma + col0))[jv];
          f4 b = ((const f4*)(beta + col0))[jv];
          v = (v - mean) * rs * g + b;
        }
        orow[jv] = v;
      }
    }
  }
}

// ---------------------------------------------------------------------------
// Fused FFN + residual + LayerNorm: out = LN(X + relu(X@W1+b1)@W2 + b2)
// 128 -> 256 -> 128; hidden kept in LDS as f16 (two 128-col passes).
// ---------------------------------------------------------------------------
__global__ __launch_bounds__(256, 1)
void ffn_fused_kernel(const float* __restrict__ X,
                      const float* __restrict__ W1,
                      const float* __restrict__ b1,
                      const float* __restrict__ W2,
                      const float* __restrict__ b2,
                      const float* __restrict__ gamma,
                      const float* __restrict__ beta,
                      float* __restrict__ out, int M) {
  extern __shared__ char smem[];
  _Float16* As = (_Float16*)smem;                       // [128][LDA_S]
  _Float16* Bs = As + 128 * LDA_S;                      // [128][LDB_S]
  char* upool  = (char*)(Bs + 128 * LDB_S);
  _Float16* Ts = (_Float16*)upool;                      // [128][LDT_S] f16 hidden
  float*    Cs = (float*)upool;                         // [128][LDC_S] aliases Ts

  const int tid  = threadIdx.x;
  const int lane = tid & 31;
  const int wave = tid >> 5;
  const int wm   = wave & 3;
  const int wn   = wave >> 2;
  const int row0 = blockIdx.x * 128;
  const int halfsel = lane >> 4;
  const int cl = lane & 15;

  // ------------- stage 1: T = relu(X@W1 + b1), in two 128-col halves --------
  for (int hh = 0; hh < 2; ++hh) {
    v8f acc1[2][4];
#pragma unroll
    for (int mt = 0; mt < 2; ++mt)
#pragma unroll
      for (int nt = 0; nt < 4; ++nt)
#pragma unroll
        for (int r = 0; r < 8; ++r) acc1[mt][nt][r] = 0.0f;

    for (int k0 = 0; k0 < 128; k0 += 32) {
      stage_a_tile(X, As, row0, k0, M, 128, tid);
      stage_b_tile(W1 + (size_t)k0 * 256 + hh * 128, Bs, 256, tid);
      __syncthreads();

      v16h af[2], bf[4];
#pragma unroll
      for (int mt = 0; mt < 2; ++mt)
        af[mt] = load_a_frag(As + (wm * 32 + mt * 16) * LDA_S, LDA_S, lane);
#pragma unroll
      for (int nt = 0; nt < 4; ++nt)
        bf[nt] = load_b_frag(Bs + (wn * 64 + nt * 16) * LDB_S, LDB_S, lane);
#pragma unroll
      for (int mt = 0; mt < 2; ++mt)
#pragma unroll
        for (int nt = 0; nt < 4; ++nt)
          acc1[mt][nt] = wmma16(af[mt], bf[nt], acc1[mt][nt]);
      __syncthreads();
    }

    // bias + relu -> Ts half
#pragma unroll
    for (int mt = 0; mt < 2; ++mt)
#pragma unroll
      for (int nt = 0; nt < 4; ++nt) {
        int rb = wm * 32 + mt * 16 + halfsel * 8;
        int cb = hh * 128 + wn * 64 + nt * 16 + cl;
        float bb = b1[cb];
#pragma unroll
        for (int r = 0; r < 8; ++r) {
          float v = fmaxf(acc1[mt][nt][r] + bb, 0.0f);
          Ts[(rb + r) * LDT_S + cb] = (_Float16)v;
        }
      }
  }

  // ------------- stage 2: Y = T @ W2, 128x128, K=256 ------------------------
  v8f acc2[2][4];
#pragma unroll
  for (int mt = 0; mt < 2; ++mt)
#pragma unroll
    for (int nt = 0; nt < 4; ++nt)
#pragma unroll
      for (int r = 0; r < 8; ++r) acc2[mt][nt][r] = 0.0f;

  for (int k0 = 0; k0 < 256; k0 += 32) {
    __syncthreads();  // prior Bs reads (and stage-1 Ts writes) complete
    stage_b_tile(W2 + (size_t)k0 * 128, Bs, 128, tid);
    __syncthreads();

    v16h af[2], bf[4];
#pragma unroll
    for (int mt = 0; mt < 2; ++mt)
      af[mt] = load_a_frag(Ts + (wm * 32 + mt * 16) * LDT_S + k0, LDT_S, lane);
#pragma unroll
    for (int nt = 0; nt < 4; ++nt)
      bf[nt] = load_b_frag(Bs + (wn * 64 + nt * 16) * LDB_S, LDB_S, lane);
#pragma unroll
    for (int mt = 0; mt < 2; ++mt)
#pragma unroll
      for (int nt = 0; nt < 4; ++nt)
        acc2[mt][nt] = wmma16(af[mt], bf[nt], acc2[mt][nt]);
  }
  __syncthreads();  // Ts reads done; alias region as Cs

#pragma unroll
  for (int mt = 0; mt < 2; ++mt)
#pragma unroll
    for (int nt = 0; nt < 4; ++nt) {
      int rb = wm * 32 + mt * 16 + halfsel * 8;
      int cb = wn * 64 + nt * 16 + cl;
#pragma unroll
      for (int r = 0; r < 8; ++r) Cs[(rb + r) * LDC_S + cb] = acc2[mt][nt][r];
    }
  __syncthreads();

  // epilogue: residual + bias + LayerNorm (vectorized)
  if (tid < 128) {
    int gr = row0 + tid;
    if (gr < M) {
      const f4* crow = (const f4*)(Cs + tid * LDC_S);
      const f4* b2v  = (const f4*)b2;
      const f4* xrow = (const f4*)(X + (size_t)gr * 128);
      f4*       orow = (f4*)(out + (size_t)gr * 128);
      float s = 0.0f, s2 = 0.0f;
#pragma unroll 4
      for (int jv = 0; jv < 32; ++jv) {
        f4 v = crow[jv] + b2v[jv] + xrow[jv];
        s  += v.x + v.y + v.z + v.w;
        s2 += v.x * v.x + v.y * v.y + v.z * v.z + v.w * v.w;
      }
      float mean = s * (1.0f / 128.0f);
      float var  = s2 * (1.0f / 128.0f) - mean * mean;
      float rs   = rsqrtf(var + 1e-5f);
#pragma unroll 4
      for (int jv = 0; jv < 32; ++jv) {
        f4 v = crow[jv] + b2v[jv] + xrow[jv];
        f4 g = ((const f4*)gamma)[jv];
        f4 b = ((const f4*)beta)[jv];
        orow[jv] = (v - mean) * rs * g + b;
      }
    }
  }
}

// ---------------------------------------------------------------------------
// Edge attention: one wave32 per edge.
// ---------------------------------------------------------------------------
__global__ void edge_attn_kernel(const int* __restrict__ src,
                                 const int* __restrict__ dst,
                                 const float* __restrict__ Q,
                                 const float* __restrict__ Kf,
                                 const float* __restrict__ Vf,
                                 float* __restrict__ P,   // in: pe, out: e_out
                                 float* __restrict__ wV,
                                 float* __restrict__ z, int E) {
  int gwave  = (blockIdx.x * blockDim.x + threadIdx.x) >> 5;
  int nwaves = (gridDim.x * blockDim.x) >> 5;
  int lane   = threadIdx.x & 31;

  for (int e = gwave; e < E; e += nwaves) {
    int s = src[e], d = dst[e];
    size_t sb = (size_t)s * 128, db = (size_t)d * 128, eb = (size_t)e * 128;
    float sexp[4];
#pragma unroll
    for (int c = 0; c < 4; ++c) {
      int j = lane + 32 * c;
      float sc = Kf[sb + j] * Q[db + j] * 0.25f * P[eb + j];
      P[eb + j] = sc;  // e_out
      float sum = sc;
      sum += __shfl_xor(sum, 1, 32);
      sum += __shfl_xor(sum, 2, 32);
      sum += __shfl_xor(sum, 4, 32);
      sum += __shfl_xor(sum, 8, 32);
      sexp[c] = expf(fminf(fmaxf(sum, -5.0f), 5.0f));
      atomicAdd(&wV[db + j], Vf[sb + j] * sexp[c]);
    }
    if ((lane & 15) == 0) {
      int hh = lane >> 4;
#pragma unroll
      for (int c = 0; c < 4; ++c)
        atomicAdd(&z[(size_t)d * 8 + c * 2 + hh], sexp[c]);
    }
  }
}

__global__ void hattn_finalize_kernel(const float* __restrict__ wV,
                                      const float* __restrict__ z,
                                      float* __restrict__ h_attn, long long n) {
  long long i = (long long)blockIdx.x * blockDim.x + threadIdx.x;
  if (i < n) {
    long long row = i >> 7;
    int head = (int)((i >> 4) & 7);
    h_attn[i] = wV[i] / (z[row * 8 + head] + 1e-6f);
  }
}

// ---------------------------------------------------------------------------
// Host-side orchestration
// ---------------------------------------------------------------------------
extern "C" void kernel_launch(void* const* d_in, const int* in_sizes, int n_in,
                              void* d_out, int out_size, void* d_ws, size_t ws_size,
                              hipStream_t stream) {
  const float* h   = (const float*)d_in[0];
  const float* e   = (const float*)d_in[1];
  const int*   src = (const int*)d_in[2];
  const int*   dst = (const int*)d_in[3];
  const float* WQ  = (const float*)d_in[4];
  const float* WK  = (const float*)d_in[5];
  const float* WV  = (const float*)d_in[6];
  const float* We  = (const float*)d_in[7];
  const float* WOh = (const float*)d_in[8];
  const float* bOh = (const float*)d_in[9];
  const float* WOe = (const float*)d_in[10];
  const float* bOe = (const float*)d_in[11];
  const float* g1h = (const float*)d_in[12];
  const float* b1h = (const float*)d_in[13];
  const float* g1e = (const float*)d_in[14];
  const float* b1e = (const float*)d_in[15];
  const float* Wh1 = (const float*)d_in[16];
  const float* bh1 = (const float*)d_in[17];
  const float* Wh2 = (const float*)d_in[18];
  const float* bh2 = (const float*)d_in[19];
  const float* We1 = (const float*)d_in[20];
  const float* be1 = (const float*)d_in[21];
  const float* We2 = (const float*)d_in[22];
  const float* be2 = (const float*)d_in[23];
  const float* g2h = (const float*)d_in[24];
  const float* b2h = (const float*)d_in[25];
  const float* g2e = (const float*)d_in[26];
  const float* b2e = (const float*)d_in[27];

  const int Nn = in_sizes[0] / 128;   // nodes
  const int Ee = in_sizes[2];         // edges
  const size_t ND = (size_t)Nn * 128;
  const size_t ED = (size_t)Ee * 128;

  float* w      = (float*)d_ws;
  float* Qb     = w;            w += ND;
  float* Kb     = w;            w += ND;
  float* Vb     = w;            w += ND;
  float* wV     = w;            w += ND;
  float* zb     = w;            w += (size_t)Nn * 8;
  float* h_attn = w;            w += ND;
  float* h1     = w;            w += ND;
  float* P      = w;            w += ED;   // pe, then e_out in place
  float* e1     = w;            w += ED;

  float* out_h2 = (float*)d_out;
  float* out_e2 = out_h2 + ND;

  const int nodeBlocks = (Nn + 127) / 128;
  const int edgeBlocks = (Ee + 127) / 128;
  constexpr int LNF = FLAG_BIAS | FLAG_RES | FLAG_LN;

  zero_f32_kernel<<<1024, 256, 0, stream>>>(wV, (long long)(ND + (size_t)Nn * 8));

  // projections
  gemm_wmma_kernel<0><<<dim3(nodeBlocks, 1), 256, GEMM_SMEM_BYTES, stream>>>(
      h, WQ, nullptr, nullptr, nullptr, nullptr, Qb, Nn, 128, 128);
  gemm_wmma_kernel<0><<<dim3(nodeBlocks, 1), 256, GEMM_SMEM_BYTES, stream>>>(
      h, WK, nullptr, nullptr, nullptr, nullptr, Kb, Nn, 128, 128);
  gemm_wmma_kernel<0><<<dim3(nodeBlocks, 1), 256, GEMM_SMEM_BYTES, stream>>>(
      h, WV, nullptr, nullptr, nullptr, nullptr, Vb, Nn, 128, 128);
  gemm_wmma_kernel<0><<<dim3(edgeBlocks, 1), 256, GEMM_SMEM_BYTES, stream>>>(
      e, We, nullptr, nullptr, nullptr, nullptr, P, Ee, 128, 128);

  // edge attention scatter + softmax segment sums
  edge_attn_kernel<<<4096, 256, 0, stream>>>(src, dst, Qb, Kb, Vb, P, wV, zb, Ee);
  hattn_finalize_kernel<<<(int)((ND + 255) / 256), 256, 0, stream>>>(
      wV, zb, h_attn, (long long)ND);

  // output projections + residual + LN
  gemm_wmma_kernel<LNF><<<dim3(nodeBlocks, 1), 256, GEMM_SMEM_BYTES, stream>>>(
      h_attn, WOh, bOh, h, g1h, b1h, h1, Nn, 128, 128);
  gemm_wmma_kernel<LNF><<<dim3(edgeBlocks, 1), 256, GEMM_SMEM_BYTES, stream>>>(
      P, WOe, bOe, e, g1e, b1e, e1, Ee, 128, 128);

  // fused FFN + residual + LN -> final outputs
  ffn_fused_kernel<<<nodeBlocks, 256, FFN_SMEM_BYTES, stream>>>(
      h1, Wh1, bh1, Wh2, bh2, g2h, b2h, out_h2, Nn);
  ffn_fused_kernel<<<edgeBlocks, 256, FFN_SMEM_BYTES, stream>>>(
      e1, We1, be1, We2, be2, g2e, b2e, out_e2, Ee);
}
